// CharRNN_86792699118071
// MI455X (gfx1250) — matrix operand order repaired and checked by reading, not compiled, not run on hardware
//
#include <hip/hip_runtime.h>

// ---------------------------------------------------------------------------
// CharRNN forward for MI455X (gfx1250, wave32, WMMA + TDM).
//   SEQLEN=1024, BATCH=128, HIDDEN=512, EMBED=128, VOCAB=128
// Plan:
//   1) convert weights (W_ih, W_hh, W_ho, embedding) f32 -> bf16 in workspace
//   2) input_proj: X[t][b][h] = b_h[h] + emb(x[b][t]) @ W_ih^T   (WMMA bf16)
//   3) 1024x rnn_step: X[t] = tanh(X[t] + X[t-1] @ W_hh^T)       (WMMA bf16,
//      A-tile staged to LDS via tensor_load_to_lds, hw v_tanh_f32)
//   4) output_proj: logits[b][t][v] = b_o[v] + X[t][b] @ W_ho^T  (WMMA bf16)
//   5) copy final hidden X[1023] -> tail of d_out
// ---------------------------------------------------------------------------

#define SEQLEN 1024
#define BATCH  128
#define HIDDEN 512
#define EMBED  128
#define VOCAB  128

typedef __attribute__((ext_vector_type(16))) __bf16    v16bf;
typedef __attribute__((ext_vector_type(8)))  float     v8f;
typedef __attribute__((ext_vector_type(4)))  unsigned  v4u;
typedef __attribute__((ext_vector_type(8)))  int       v8i;
typedef __attribute__((ext_vector_type(4)))  int       v4i;

union ABfrag { unsigned u[8]; v16bf v; };

// round-to-nearest-even f32 -> bf16, packed pair into one dword
__device__ __forceinline__ unsigned pack2_bf16(float lo, float hi) {
  unsigned a = __builtin_bit_cast(unsigned, lo);
  unsigned b = __builtin_bit_cast(unsigned, hi);
  a = (a + 0x7FFFu + ((a >> 16) & 1u)) >> 16;
  b = (b + 0x7FFFu + ((b >> 16) & 1u)) >> 16;
  return (a & 0xFFFFu) | (b << 16);
}

// A-matrix (16x32 bf16) K index for VGPR i, lane-half h  (ISA 7.12.2)
__device__ __forceinline__ int kmapA(int i, int h) {
  return ((i < 4) ? (2 * i) : (16 + 2 * (i - 4))) + 8 * h;
}
// B-matrix (32x16 bf16) K index for VGPR i, lane-half h
__device__ __forceinline__ int kmapB(int i, int h) {
  return ((i < 4) ? (2 * i) : (16 + 2 * (i - 4))) + 8 * h;
}

// gfx1250 hardware tanh (transcendental); v_nop covers the TRANS result hazard
__device__ __forceinline__ float fast_tanh(float x) {
  float y;
  asm volatile("v_tanh_f32 %0, %1\n\tv_nop" : "=v"(y) : "v"(x));
  return y;
}

// ---------------------------------------------------------------------------
__global__ void f32_to_bf16_kernel(const float* __restrict__ src,
                                   unsigned short* __restrict__ dst, int n) {
  int i = blockIdx.x * blockDim.x + threadIdx.x;
  if (i < n) {
    unsigned a = __builtin_bit_cast(unsigned, src[i]);
    a = (a + 0x7FFFu + ((a >> 16) & 1u)) >> 16;
    dst[i] = (unsigned short)a;
  }
}

// ---------------------------------------------------------------------------
// X[t*BATCH+b][h] = b_h[h] + sum_e emb[x[b][t]][e] * W_ih[h][e]
// one wave -> 16 rows x 64 cols (4 WMMA tiles, shared A)
__global__ void __launch_bounds__(256) input_proj_kernel(
    const long long* __restrict__ x,          // [BATCH][SEQLEN] int64
    const unsigned short* __restrict__ embB,  // [VOCAB][EMBED] bf16
    const unsigned short* __restrict__ WihB,  // [HIDDEN][EMBED] bf16
    const float* __restrict__ b_h,            // [HIDDEN]
    float* __restrict__ X)                    // [SEQLEN][BATCH][HIDDEN]
{
  const int NSTRIP = HIDDEN / 64;  // 8
  int w      = blockIdx.x * (blockDim.x >> 5) + (threadIdx.x >> 5);
  int mtile  = w / NSTRIP;
  int nstrip = w % NSTRIP;
  int lane = threadIdx.x & 31;
  int half = lane >> 4;
  int idx  = lane & 15;

  // this lane's A row: flat row = t*BATCH + b
  int rowA = mtile * 16 + idx;
  int tA = rowA >> 7;        // / BATCH
  int bA = rowA & (BATCH - 1);
  int tok = (int)x[(size_t)bA * SEQLEN + tA];
  const unsigned short* arow = embB + (size_t)tok * EMBED;

  v8f acc[4] = {};
  for (int kc = 0; kc < EMBED; kc += 32) {
    ABfrag A;
#pragma unroll
    for (int i = 0; i < 8; ++i)
      A.u[i] = *(const unsigned*)(arow + kc + kmapA(i, half));
#pragma unroll
    for (int s = 0; s < 4; ++s) {
      int n = nstrip * 64 + s * 16 + idx;
      const unsigned short* brow = WihB + (size_t)n * EMBED;
      ABfrag B;
#pragma unroll
      for (int i = 0; i < 8; ++i)
        B.u[i] = *(const unsigned*)(brow + kc + kmapB(i, half));
      acc[s] = __builtin_amdgcn_wmma_f32_16x16x32_bf16(
          false, A.v, false, B.v, (short)0, acc[s], false, false);
    }
  }
#pragma unroll
  for (int s = 0; s < 4; ++s) {
    int n = nstrip * 64 + s * 16 + idx;
    float bias = b_h[n];
#pragma unroll
    for (int r = 0; r < 8; ++r) {
      int row = mtile * 16 + r + 8 * half;
      X[(size_t)row * HIDDEN + n] = acc[s][r] + bias;
    }
  }
}

// ---------------------------------------------------------------------------
// Xt[b][h] = tanh(Xt[b][h] + sum_k Hprev[b][k] * W_hh[h][k])   (in-place)
// grid: 8 blocks (one per 16-row M tile), 8 waves = 8 x 64-col N strips.
// A tile (16x512 f32, 32KB) is staged into LDS once per block via the
// Tensor Data Mover, then each wave reads its fragments from LDS.
__global__ void __launch_bounds__(256) rnn_step_kernel(
    const float* __restrict__ Hprev,          // [BATCH][HIDDEN] f32
    const unsigned short* __restrict__ WhhB,  // [HIDDEN][HIDDEN] bf16
    float* __restrict__ Xt)                   // [BATCH][HIDDEN] f32
{
  __shared__ float As[16 * HIDDEN];  // 32 KB A tile (f32)

  int mtile  = blockIdx.x;               // 0..7 : all waves share this M tile
  int nstrip = threadIdx.x >> 5;         // 0..7
  int lane = threadIdx.x & 31;
  int half = lane >> 4;
  int idx  = lane & 15;

  // ---- TDM: Hprev[mtile*16 .. +16][0..512] f32 -> LDS --------------------
  if (threadIdx.x < 32) {
    unsigned lds = (unsigned)(size_t)(&As[0]);
    unsigned long long ga =
        (unsigned long long)(size_t)(Hprev + (size_t)mtile * 16 * HIDDEN);
    // D# group0: count=1 | lds_addr | global_addr | type=2   (ISA 8.3)
    v4u g0 = { 1u, lds, (unsigned)(ga & 0xffffffffu),
               (unsigned)((ga >> 32) & 0x01ffffffu) | (2u << 30) };
    // D# group1: data_size=4B, tensor 512x128, tile 512x16, stride 512 (ISA 8.4)
    v8i g1 = { (int)0x00020000u,        // wg_mask=0, data_size=2 (4B)
               (int)(512u << 16),       // tensor_dim0[15:0] = 512
               (int)(128u << 16),       // tensor_dim1[15:0] = 128
               (int)(512u << 16),       // tile_dim0 = 512
               16,                      // tile_dim1 = 16
               512,                     // tensor_dim0_stride = 512
               0, 0 };
    v4i gz4 = { 0, 0, 0, 0 };
    v8i gz8 = { 0, 0, 0, 0, 0, 0, 0, 0 };
    __builtin_amdgcn_tensor_load_to_lds(g0, g1, gz4, gz4, gz8, 0);
    __builtin_amdgcn_s_wait_tensorcnt(0);
  }
  __syncthreads();

  v8f acc[4] = {};
  for (int kc = 0; kc < HIDDEN; kc += 32) {
    ABfrag A;
#pragma unroll
    for (int i = 0; i < 8; ++i) {
      int kk = kc + kmapA(i, half);
      float2 p = *(const float2*)(&As[idx * HIDDEN + kk]);  // LDS read
      A.u[i] = pack2_bf16(p.x, p.y);
    }
#pragma unroll
    for (int s = 0; s < 4; ++s) {
      int n = nstrip * 64 + s * 16 + idx;
      const unsigned short* brow = WhhB + (size_t)n * HIDDEN;
      ABfrag B;
#pragma unroll
      for (int i = 0; i < 8; ++i)
        B.u[i] = *(const unsigned*)(brow + kc + kmapB(i, half));
      acc[s] = __builtin_amdgcn_wmma_f32_16x16x32_bf16(
          false, A.v, false, B.v, (short)0, acc[s], false, false);
    }
  }
#pragma unroll
  for (int s = 0; s < 4; ++s) {
    int n = nstrip * 64 + s * 16 + idx;
#pragma unroll
    for (int r = 0; r < 8; ++r) {
      int row = mtile * 16 + r + 8 * half;
      size_t o = (size_t)row * HIDDEN + n;
      Xt[o] = fast_tanh(acc[s][r] + Xt[o]);
    }
  }
}

// ---------------------------------------------------------------------------
// out[b*SEQLEN+t][v] = b_o[v] + sum_h H[t][b][h] * W_ho[v][h]
__global__ void __launch_bounds__(256) output_proj_kernel(
    const float* __restrict__ H,              // [SEQLEN][BATCH][HIDDEN] f32
    const unsigned short* __restrict__ WhoB,  // [VOCAB][HIDDEN] bf16
    const float* __restrict__ b_o,            // [VOCAB]
    float* __restrict__ out)                  // [BATCH][SEQLEN][VOCAB]
{
  const int NSTRIP = VOCAB / 64;  // 2
  int w      = blockIdx.x * (blockDim.x >> 5) + (threadIdx.x >> 5);
  int mtile  = w / NSTRIP;
  int nstrip = w % NSTRIP;
  int lane = threadIdx.x & 31;
  int half = lane >> 4;
  int idx  = lane & 15;

  int rowA = mtile * 16 + idx;      // flat b*SEQLEN + t
  int bA = rowA >> 10;              // / SEQLEN
  int tA = rowA & (SEQLEN - 1);
  const float* arow = H + ((size_t)tA * BATCH + bA) * HIDDEN;

  v8f acc[4] = {};
  for (int kc = 0; kc < HIDDEN; kc += 32) {
    ABfrag A;
#pragma unroll
    for (int i = 0; i < 8; ++i) {
      int kk = kc + kmapA(i, half);
      float2 p = *(const float2*)(arow + kk);
      A.u[i] = pack2_bf16(p.x, p.y);
    }
#pragma unroll
    for (int s = 0; s < 4; ++s) {
      int n = nstrip * 64 + s * 16 + idx;
      const unsigned short* brow = WhoB + (size_t)n * HIDDEN;
      ABfrag B;
#pragma unroll
      for (int i = 0; i < 8; ++i)
        B.u[i] = *(const unsigned*)(brow + kc + kmapB(i, half));
      acc[s] = __builtin_amdgcn_wmma_f32_16x16x32_bf16(
          false, A.v, false, B.v, (short)0, acc[s], false, false);
    }
  }
#pragma unroll
  for (int s = 0; s < 4; ++s) {
    int n = nstrip * 64 + s * 16 + idx;
    float bias = b_o[n];
#pragma unroll
    for (int r = 0; r < 8; ++r) {
      int row = mtile * 16 + r + 8 * half;
      out[(size_t)row * VOCAB + n] = acc[s][r] + bias;
    }
  }
}

// ---------------------------------------------------------------------------
__global__ void copy_f32_kernel(const float* __restrict__ src,
                                float* __restrict__ dst, int n) {
  int i = blockIdx.x * blockDim.x + threadIdx.x;
  if (i < n) dst[i] = src[i];
}

// ---------------------------------------------------------------------------
extern "C" void kernel_launch(void* const* d_in, const int* in_sizes, int n_in,
                              void* d_out, int out_size, void* d_ws, size_t ws_size,
                              hipStream_t stream) {
  const long long* x      = (const long long*)d_in[0];  // int64 tokens
  const float* hidden0    = (const float*)d_in[1];
  const float* embedding  = (const float*)d_in[2];
  const float* W_ih       = (const float*)d_in[3];
  const float* W_hh       = (const float*)d_in[4];
  const float* b_h        = (const float*)d_in[5];
  const float* W_ho       = (const float*)d_in[6];
  const float* b_o        = (const float*)d_in[7];

  char* ws = (char*)d_ws;
  float* X = (float*)ws;                                   // 256 MB activations
  size_t off = (size_t)SEQLEN * BATCH * HIDDEN * sizeof(float);
  unsigned short* WhhB = (unsigned short*)(ws + off); off += (size_t)HIDDEN * HIDDEN * 2;
  unsigned short* WihB = (unsigned short*)(ws + off); off += (size_t)HIDDEN * EMBED  * 2;
  unsigned short* WhoB = (unsigned short*)(ws + off); off += (size_t)VOCAB  * HIDDEN * 2;
  unsigned short* embB = (unsigned short*)(ws + off); off += (size_t)VOCAB  * EMBED  * 2;

  // 1) weight conversion f32 -> bf16
  f32_to_bf16_kernel<<<(HIDDEN * HIDDEN + 255) / 256, 256, 0, stream>>>(W_hh, WhhB, HIDDEN * HIDDEN);
  f32_to_bf16_kernel<<<(HIDDEN * EMBED  + 255) / 256, 256, 0, stream>>>(W_ih, WihB, HIDDEN * EMBED);
  f32_to_bf16_kernel<<<(VOCAB  * HIDDEN + 255) / 256, 256, 0, stream>>>(W_ho, WhoB, VOCAB * HIDDEN);
  f32_to_bf16_kernel<<<(VOCAB  * EMBED  + 255) / 256, 256, 0, stream>>>(embedding, embB, VOCAB * EMBED);

  // 2) input projection: 8192 M-tiles x 8 N-strips = 65536 waves = 8192 blocks
  {
    int waves = (SEQLEN * BATCH / 16) * (HIDDEN / 64);
    input_proj_kernel<<<waves / 8, 256, 0, stream>>>(x, embB, WihB, b_h, X);
  }

  // 3) recurrence: 1024 chained step kernels (8 blocks, one M tile each)
  for (int t = 0; t < SEQLEN; ++t) {
    const float* Hprev = (t == 0) ? hidden0 : (X + (size_t)(t - 1) * BATCH * HIDDEN);
    rnn_step_kernel<<<8, 256, 0, stream>>>(Hprev, WhhB, X + (size_t)t * BATCH * HIDDEN);
  }

  // 4) output projection: 8192 M-tiles x 2 N-strips = 16384 waves = 2048 blocks
  {
    int waves = (BATCH * SEQLEN / 16) * (VOCAB / 64);
    output_proj_kernel<<<waves / 8, 256, 0, stream>>>(X, WhoB, b_o, (float*)d_out);
  }

  // 5) final hidden -> tail of d_out
  copy_f32_kernel<<<(BATCH * HIDDEN + 255) / 256, 256, 0, stream>>>(
      X + (size_t)(SEQLEN - 1) * BATCH * HIDDEN,
      (float*)d_out + (size_t)BATCH * SEQLEN * VOCAB, BATCH * HIDDEN);
}
